// OmgContrastiveLoss_37031208026193
// MI455X (gfx1250) — matrix operand order repaired and checked by reading, not compiled, not run on hardware
//
#include <hip/hip_runtime.h>

typedef __attribute__((ext_vector_type(16))) __bf16 v16bf;
typedef __attribute__((ext_vector_type(8)))  float  v8f;

#define NN 8192
#define DD 128
#define GX 8                       // column partitions of the Gram sweep
#define COLS_PER_BLK (NN / GX)     // 1024 columns per block

static __device__ __forceinline__ unsigned short f32_to_bf16(float f) {
  union { float f; unsigned int u; } x; x.f = f;
  unsigned int u = x.u;
  unsigned int r = u + 0x7FFFu + ((u >> 16) & 1u);   // round-to-nearest-even
  return (unsigned short)(r >> 16);
}

// Raw v_sqrt_f32: inputs are distances^2 of unit vectors, range [0,4] —
// the precise-sqrt denormal rescale path can never trigger, skip it.
static __device__ __forceinline__ float fast_sqrt(float x) {
  return __builtin_amdgcn_sqrtf(x);
}

// --- Kernel 1: normalize rows, emit bf16 p, sq[i], negd[i] -----------------
__global__ void __launch_bounds__(256)
prep_kernel(const float* __restrict__ pred, const float* __restrict__ neg,
            unsigned short* __restrict__ pB, float* __restrict__ sq,
            float* __restrict__ negd) {
  const int wave = threadIdx.x >> 5;
  const int lane = threadIdx.x & 31;
  const int row  = blockIdx.x * 8 + wave;

  float4 pr = ((const float4*)(pred + (size_t)row * DD))[lane];
  float4 ng = ((const float4*)(neg  + (size_t)row * DD))[lane];

  float sp = pr.x*pr.x + pr.y*pr.y + pr.z*pr.z + pr.w*pr.w;
  float sn = ng.x*ng.x + ng.y*ng.y + ng.z*ng.z + ng.w*ng.w;
  #pragma unroll
  for (int m = 16; m >= 1; m >>= 1) {
    sp += __shfl_xor(sp, m, 32);
    sn += __shfl_xor(sn, m, 32);
  }
  const float invp = 1.0f / fmaxf(sqrtf(sp), 1e-12f);
  const float invn = 1.0f / fmaxf(sqrtf(sn), 1e-12f);

  float4 p, n;
  p.x = pr.x*invp; p.y = pr.y*invp; p.z = pr.z*invp; p.w = pr.w*invp;
  n.x = ng.x*invn; n.y = ng.y*invn; n.z = ng.z*invn; n.w = ng.w*invn;

  float dx = p.x-n.x, dy = p.y-n.y, dz = p.z-n.z, dw = p.w-n.w;
  float sd = dx*dx + dy*dy + dz*dz + dw*dw;
  #pragma unroll
  for (int m = 16; m >= 1; m >>= 1) sd += __shfl_xor(sd, m, 32);

  if (lane == 0) {
    sq[row]   = sp * invp * invp;                 // sum(p*p) after normalize
    negd[row] = (sd > 0.0f) ? sqrtf(sd) : 0.0f;   // safe_sqrt
  }
  union { unsigned short s[4]; uint2 u; } pk;
  pk.s[0] = f32_to_bf16(p.x); pk.s[1] = f32_to_bf16(p.y);
  pk.s[2] = f32_to_bf16(p.z); pk.s[3] = f32_to_bf16(p.w);
  ((uint2*)(pB + (size_t)row * DD))[lane] = pk.u;
}

// --- Kernel 2: WMMA Gram tiles fused with masked distance reduction --------
__global__ void __launch_bounds__(256)
gram_kernel(const unsigned short* __restrict__ pB, const float* __restrict__ sq,
            const float* __restrict__ mask, float* __restrict__ pos_sum,
            float* __restrict__ cnt) {
  const int wave    = threadIdx.x >> 5;
  const int lane    = threadIdx.x & 31;
  const int rowTile = wave & 3;          // 4 row tiles  -> 64 rows / block
  const int colTile = wave >> 2;         // 2 col tiles  -> 32 cols / step
  const int half    = lane >> 4;
  const int l15     = lane & 15;
  const int rowBase = blockIdx.y * 64 + rowTile * 16;

  // A fragment (ISA 16-bit A 16x32): lane half 0 -> K {c*32+0..7, c*32+16..23},
  // lane half 1 -> K {c*32+8..15, c*32+24..31}; row = rowBase + l15.
  v16bf a[4];
  {
    const unsigned short* ap = pB + (size_t)(rowBase + l15) * DD;
    #pragma unroll
    for (int c = 0; c < 4; ++c) {
      union { uint4 q[2]; v16bf v; } ld;
      ld.q[0] = *(const uint4*)(ap + c*32 + half*8);
      ld.q[1] = *(const uint4*)(ap + c*32 + 16 + half*8);
      a[c] = ld.v;
    }
  }
  float sqr[8];
  #pragma unroll
  for (int r = 0; r < 8; ++r) sqr[r] = sq[rowBase + half*8 + r];

  float ppos[8], pcnt[8];
  #pragma unroll
  for (int r = 0; r < 8; ++r) { ppos[r] = 0.0f; pcnt[r] = 0.0f; }

  const int colStart = blockIdx.x * COLS_PER_BLK;
  for (int cb = 0; cb < COLS_PER_BLK; cb += 32) {
    const int colBase = colStart + cb + colTile * 16;
    const int bcol    = colBase + l15;
    // B fragment (32x16 bf16): lane half 0 holds K {0..7,16..23}, half 1 K {8..15,24..31},
    // column = colBase + l15 -> two 16B runs of row `bcol` of p.
    const unsigned short* bp = pB + (size_t)bcol * DD;
    v8f acc = {};
    #pragma unroll
    for (int c = 0; c < 4; ++c) {
      union { uint4 q[2]; v16bf v; } ld;
      ld.q[0] = *(const uint4*)(bp + c*32 + half*8);
      ld.q[1] = *(const uint4*)(bp + c*32 + 16 + half*8);
      acc = __builtin_amdgcn_wmma_f32_16x16x32_bf16(
          false, a[c], false, ld.v, (short)0, acc, false, false);
    }
    const float  sqc  = sq[bcol];
    const float* mrow = mask + (size_t)(rowBase + half*8) * NN + bcol;
    #pragma unroll
    for (int r = 0; r < 8; ++r) {        // C layout: VGPR r -> M = half*8 + r
      float g    = acc[r];
      float d2   = fmaxf(sqr[r] + sqc - 2.0f * g, 0.0f);
      float dist = fast_sqrt(d2);        // safe_sqrt: d2==0 -> 0
      // Mask is 256 MB streamed exactly once (> 192 MB L2): non-temporal so
      // it doesn't evict the hot 2 MB bf16 operand / sq arrays from L2.
      float mv   = __builtin_nontemporal_load(mrow + (size_t)r * NN);
      ppos[r] += dist * mv;
      pcnt[r] += mv;
    }
  }
  // Reduce the 16 lanes holding one output row, then one atomic per row.
  #pragma unroll
  for (int r = 0; r < 8; ++r) {
    float pp = ppos[r], pc = pcnt[r];
    #pragma unroll
    for (int m = 8; m >= 1; m >>= 1) {
      pp += __shfl_xor(pp, m, 32);
      pc += __shfl_xor(pc, m, 32);
    }
    if (l15 == 0) {
      const int row = rowBase + half*8 + r;
      atomicAdd(&pos_sum[row], pp);
      atomicAdd(&cnt[row], pc);
    }
  }
}

// --- Kernel 3: scalar mean --------------------------------------------------
__global__ void __launch_bounds__(256)
finalize_kernel(const float* __restrict__ pos_sum, const float* __restrict__ cnt,
                const float* __restrict__ negd, float* __restrict__ out) {
  __shared__ float red[256];
  float s = 0.0f;
  for (int i = threadIdx.x; i < NN; i += 256)
    s += pos_sum[i] / fmaxf(cnt[i], 1.0f) - negd[i];
  red[threadIdx.x] = s;
  __syncthreads();
  for (int st = 128; st > 0; st >>= 1) {
    if (threadIdx.x < st) red[threadIdx.x] += red[threadIdx.x + st];
    __syncthreads();
  }
  if (threadIdx.x == 0) out[0] = red[0] / (float)NN;
}

extern "C" void kernel_launch(void* const* d_in, const int* in_sizes, int n_in,
                              void* d_out, int out_size, void* d_ws, size_t ws_size,
                              hipStream_t stream) {
  const float* pred = (const float*)d_in[0];
  const float* mask = (const float*)d_in[1];
  const float* neg  = (const float*)d_in[2];

  char* ws = (char*)d_ws;
  unsigned short* pB      = (unsigned short*)ws;                 // 2 MB bf16 p
  float*          sq      = (float*)(ws + 2097152);              // 32 KB
  float*          negd    = (float*)(ws + 2097152 + 32768);      // 32 KB
  float*          pos_sum = (float*)(ws + 2097152 + 65536);      // 32 KB
  float*          cnt     = (float*)(ws + 2097152 + 98304);      // 32 KB

  hipMemsetAsync(pos_sum, 0, 65536, stream);   // zero pos_sum + cnt (contiguous)

  prep_kernel<<<NN / 8, 256, 0, stream>>>(pred, neg, pB, sq, negd);

  dim3 grid(GX, NN / 64);
  gram_kernel<<<grid, 256, 0, stream>>>(pB, sq, mask, pos_sum, cnt);

  finalize_kernel<<<1, 256, 0, stream>>>(pos_sum, cnt, negd, (float*)d_out);
}
